// GQA_30296699306340
// MI455X (gfx1250) — compile-verified, matrix operand order checked by reading, not run on hardware
//
#include <hip/hip_runtime.h>

// ---------------------------------------------------------------------------
// CDNA5 (gfx1250, wave32) GQA forward:  bf16 WMMA everywhere a GEMM appears.
// ---------------------------------------------------------------------------

typedef __bf16 v16bf __attribute__((ext_vector_type(16)));
typedef float  v8f   __attribute__((ext_vector_type(8)));

struct alignas(16) U4 { unsigned int u[4]; };
union Frag16 { U4 q[2]; v16bf v; };   // 16 bf16 = 32B = one WMMA A/B operand (8 VGPRs)
union U4H8  { U4 q; __bf16 h[8]; };

constexpr int BB  = 2;
constexpr int SEQ = 2048;
constexpr int EMB = 2048;
constexpr int NH  = 16;
constexpr int NG  = 4;
constexpr int DH  = 128;
constexpr int KV  = NG * DH;          // 512

// ---------------------------------------------------------------------------
// elementwise f32 -> bf16
// ---------------------------------------------------------------------------
__global__ void cvt_bf16_kernel(const float* __restrict__ in,
                                __bf16* __restrict__ out, long n) {
  long i = (long)blockIdx.x * blockDim.x + threadIdx.x;
  long st = (long)gridDim.x * blockDim.x;
  for (; i < n; i += st) out[i] = (__bf16)in[i];
}

// ---------------------------------------------------------------------------
// RoPE + pack [B][S][Hc][DH] f32 -> [B][Hc][S][DH] bf16
// ---------------------------------------------------------------------------
__global__ void rope_pack_kernel(const float* __restrict__ in,
                                 __bf16* __restrict__ out, int Hc) {
  const long n = (long)BB * SEQ * Hc * (DH / 2);
  long i = (long)blockIdx.x * blockDim.x + threadIdx.x;
  long st = (long)gridDim.x * blockDim.x;
  for (; i < n; i += st) {
    int p = (int)(i % (DH / 2));
    long r = i / (DH / 2);
    int h = (int)(r % Hc); r /= Hc;
    int s = (int)(r % SEQ);
    int b = (int)(r / SEQ);
    // theta_p = 1000^(-2p/DH); angle = s * theta_p
    float ang = (float)s * __powf(1000.0f, -2.0f * (float)p / (float)DH);
    float c = __cosf(ang), sn = __sinf(ang);
    const float* xp = in + ((((long)b * SEQ + s) * Hc + h) * DH + 2 * p);
    float x0 = xp[0], x1 = xp[1];
    __bf16* op = out + ((((long)b * Hc + h) * SEQ + s) * DH + 2 * p);
    op[0] = (__bf16)(x0 * c - x1 * sn);
    op[1] = (__bf16)(x1 * c + x0 * sn);
  }
}

// ---------------------------------------------------------------------------
// pack V [B][S][G][DH] f32 -> [B][G][S][DH] bf16
// ---------------------------------------------------------------------------
__global__ void pack_v_kernel(const float* __restrict__ in,
                              __bf16* __restrict__ out) {
  const long n = (long)BB * SEQ * NG * DH;
  long i = (long)blockIdx.x * blockDim.x + threadIdx.x;
  long st = (long)gridDim.x * blockDim.x;
  for (; i < n; i += st) {
    int d = (int)(i % DH);
    long r = i / DH;
    int g = (int)(r % NG); r /= NG;
    int s = (int)(r % SEQ);
    int b = (int)(r / SEQ);
    out[(((long)b * NG + g) * SEQ + s) * DH + d] = (__bf16)in[i];
  }
}

// ---------------------------------------------------------------------------
// Tiled bf16 WMMA GEMM: C[Mr,N] = A[Mr,K] @ W[K,N] (+bias), all row-major.
// Block tile 64x64, BK=32, 128 threads (4 waves), each wave: 2x2 WMMA tiles.
// Software-pipelined: next K-tile's global loads overlap this tile's WMMAs.
// ---------------------------------------------------------------------------
__global__ __launch_bounds__(128)
void gemm_bf16_wmma(const __bf16* __restrict__ A, const __bf16* __restrict__ W,
                    const float* __restrict__ bias, float* __restrict__ C,
                    int Mr, int N, int K) {
  __shared__ alignas(16) __bf16 As[64 * 40];   // [row][k], pitch 40 (16B aligned)
  __shared__ alignas(16) __bf16 Bs[64 * 40];   // [n][k] transposed, pitch 40

  const int t    = threadIdx.x;
  const int lane = t & 31;
  const int wave = t >> 5;
  const int l16  = lane & 15;
  const int half = lane >> 4;          // lanes 0-15 vs 16-31 (wave32 layout)
  const int m0 = blockIdx.y * 64;
  const int n0 = blockIdx.x * 64;
  const int wm = (wave >> 1) * 32;     // wave sub-tile
  const int wn = (wave & 1) * 32;

  // per-thread staging chunk coordinates (2 x 16B for A, 2 x 16B for W)
  int arow[2], acol[2], wrow[2], wcol[2];
#pragma unroll
  for (int i = 0; i < 2; ++i) {
    int idx = t + i * 128;
    arow[i] = idx >> 2; acol[i] = (idx & 3) * 8;   // A: 64 rows x 32 k
    wrow[i] = idx >> 3; wcol[i] = (idx & 7) * 8;   // W: 32 k x 64 n
  }

  v8f acc[2][2] = {};

  // prologue: load K-tile 0 into registers
  U4 aReg[2], wReg[2];
#pragma unroll
  for (int i = 0; i < 2; ++i) {
    aReg[i] = *reinterpret_cast<const U4*>(A + (size_t)(m0 + arow[i]) * K + acol[i]);
    wReg[i] = *reinterpret_cast<const U4*>(W + (size_t)wrow[i] * N + n0 + wcol[i]);
  }

  for (int k0 = 0; k0 < K; k0 += 32) {
    // commit staged registers to LDS (single s_wait for all prior loads)
#pragma unroll
    for (int i = 0; i < 2; ++i) {
      *reinterpret_cast<U4*>(&As[arow[i] * 40 + acol[i]]) = aReg[i];
      U4H8 wv; wv.q = wReg[i];
#pragma unroll
      for (int j = 0; j < 8; ++j) Bs[(wcol[i] + j) * 40 + wrow[i]] = wv.h[j];
    }
    __syncthreads();

    // issue next K-tile's global loads NOW -> overlap with WMMA below
    if (k0 + 32 < K) {
      int kn = k0 + 32;
#pragma unroll
      for (int i = 0; i < 2; ++i) {
        aReg[i] = *reinterpret_cast<const U4*>(
            A + (size_t)(m0 + arow[i]) * K + kn + acol[i]);
        wReg[i] = *reinterpret_cast<const U4*>(
            W + (size_t)(kn + wrow[i]) * N + n0 + wcol[i]);
      }
      __builtin_prefetch(A + (size_t)(m0 + arow[0]) * K + kn + 32, 0, 3);
      __builtin_prefetch(W + (size_t)(kn + 32 + wrow[0]) * N + n0, 0, 3);
    }

    // per-lane fragment gather (16-bit A layout: e<8 -> k=half*8+e, else 16+half*8+e-8)
    Frag16 af[2], bfrag[2];
#pragma unroll
    for (int s = 0; s < 2; ++s) {
      int r = wm + s * 16 + l16;
      af[s].q[0] = *reinterpret_cast<const U4*>(&As[r * 40 + half * 8]);
      af[s].q[1] = *reinterpret_cast<const U4*>(&As[r * 40 + 16 + half * 8]);
      int n = wn + s * 16 + l16;
      bfrag[s].q[0] = *reinterpret_cast<const U4*>(&Bs[n * 40 + half * 8]);
      bfrag[s].q[1] = *reinterpret_cast<const U4*>(&Bs[n * 40 + 16 + half * 8]);
    }
#pragma unroll
    for (int i = 0; i < 2; ++i)
#pragma unroll
      for (int j = 0; j < 2; ++j)
        acc[i][j] = __builtin_amdgcn_wmma_f32_16x16x32_bf16(
            false, af[i].v, false, bfrag[j].v, (short)0, acc[i][j], false, false);
    __syncthreads();
  }

  // C layout: lane = column n (l16), VGPR v = row (v + 8*half)
#pragma unroll
  for (int i = 0; i < 2; ++i)
#pragma unroll
    for (int j = 0; j < 2; ++j) {
      int nc = n0 + wn + j * 16 + l16;
      float bv = bias ? bias[nc] : 0.0f;
#pragma unroll
      for (int v = 0; v < 8; ++v) {
        int mr = m0 + wm + i * 16 + v + half * 8;
        C[(size_t)mr * N + nc] = acc[i][j][v] + bv;
      }
    }
}

// ---------------------------------------------------------------------------
// Flash attention: one workgroup per (b, h, 64-query tile); 4 waves x 16 rows.
// Q/K frags loaded directly from global (row-major over d). V transposed in
// LDS; P re-laid out C->A via wave-private LDS + s_wait_dscnt.
// ---------------------------------------------------------------------------
__global__ __launch_bounds__(128)
void attn_fwd_wmma(const __bf16* __restrict__ Qh,   // [B][H][S][DH]
                   const __bf16* __restrict__ Kg,   // [B][G][S][DH]
                   const __bf16* __restrict__ Vg,   // [B][G][S][DH]
                   __bf16* __restrict__ O) {        // [B][S][H*DH]
  __shared__ alignas(16) __bf16 Pst[4 * 16 * 32];   // per-wave P staging
  __shared__ alignas(16) __bf16 Vt[128 * 40];       // V^T tile: [d][key], pitch 40

  const int t = threadIdx.x, lane = t & 31, wave = t >> 5;
  const int l16 = lane & 15, half = lane >> 4;
  const int b = blockIdx.z, h = blockIdx.y;
  const int g = h / (NH / NG);
  const int q0 = blockIdx.x * 64;
  const int qw = q0 + wave * 16;

  const __bf16* Qbase = Qh + (size_t)(b * NH + h) * SEQ * DH;
  const __bf16* Kbase = Kg + (size_t)(b * NG + g) * SEQ * DH;
  const __bf16* Vbase = Vg + (size_t)(b * NG + g) * SEQ * DH;

  // Q fragments for this wave's 16 rows, all of D=128 (4 x K32 chunks)
  Frag16 qf[4];
#pragma unroll
  for (int dc = 0; dc < 4; ++dc) {
    const __bf16* p = Qbase + (size_t)(qw + l16) * DH + dc * 32;
    qf[dc].q[0] = *reinterpret_cast<const U4*>(p + half * 8);
    qf[dc].q[1] = *reinterpret_cast<const U4*>(p + 16 + half * 8);
  }

  const float scale = rsqrtf((float)DH);
  v8f o[8] = {};                 // 16 x 128 f32 accumulator
  float mrow[8], lacc[8];
#pragma unroll
  for (int v = 0; v < 8; ++v) { mrow[v] = -1e30f; lacc[v] = 0.0f; }

  for (int j0 = 0; j0 < q0 + 64; j0 += 32) {      // uniform bound per workgroup
    __syncthreads();                               // Vt reuse fence

    // stage V^T: batch all 4 global loads first (one wait), then transpose-store
    U4 vreg[4];
#pragma unroll
    for (int c = 0; c < 4; ++c) {
      int idx = t + c * 128;                       // 0..511
      vreg[c] = *reinterpret_cast<const U4*>(
          Vbase + (size_t)(j0 + (idx >> 4)) * DH + (idx & 15) * 8);
    }
#pragma unroll
    for (int c = 0; c < 4; ++c) {
      int idx = t + c * 128;
      int key = idx >> 4, db = (idx & 15) * 8;
      U4H8 vv; vv.q = vreg[c];
#pragma unroll
      for (int j = 0; j < 8; ++j) Vt[(db + j) * 40 + key] = vv.h[j];
    }

    // scores S = Q K^T : 2 key sub-tiles x 4 d-chunks (K frag contiguous over d)
    v8f sf[2] = {};
#pragma unroll
    for (int sub = 0; sub < 2; ++sub)
#pragma unroll
      for (int dc = 0; dc < 4; ++dc) {
        Frag16 kf;
        const __bf16* p = Kbase + (size_t)(j0 + sub * 16 + l16) * DH + dc * 32;
        kf.q[0] = *reinterpret_cast<const U4*>(p + half * 8);
        kf.q[1] = *reinterpret_cast<const U4*>(p + 16 + half * 8);
        sf[sub] = __builtin_amdgcn_wmma_f32_16x16x32_bf16(
            false, qf[dc].v, false, kf.v, (short)0, sf[sub], false, false);
      }

    // online softmax (row m = v + 8*half; column n = key). xor<16 stays in half.
#pragma unroll
    for (int v = 0; v < 8; ++v) {
      int qrow = qw + v + half * 8;
#pragma unroll
      for (int sub = 0; sub < 2; ++sub) {
        int key = j0 + sub * 16 + l16;
        float s = sf[sub][v] * scale;
        sf[sub][v] = (key <= qrow) ? s : -1e30f;
      }
      float mx = fmaxf(sf[0][v], sf[1][v]);
#pragma unroll
      for (int off = 1; off < 16; off <<= 1)
        mx = fmaxf(mx, __shfl_xor(mx, off, 32));
      float mn = fmaxf(mrow[v], mx);
      float alpha = __expf(mrow[v] - mn);
      mrow[v] = mn;
      float p0 = __expf(sf[0][v] - mn);
      float p1 = __expf(sf[1][v] - mn);
      lacc[v] = lacc[v] * alpha + p0 + p1;
      sf[0][v] = p0; sf[1][v] = p1;
#pragma unroll
      for (int ds = 0; ds < 8; ++ds) o[ds][v] *= alpha;
    }

    // P: C-layout -> A-layout via wave-private LDS
    __bf16* P = &Pst[wave * 16 * 32];
#pragma unroll
    for (int v = 0; v < 8; ++v) {
      int m = v + half * 8;
      P[m * 32 + l16]      = (__bf16)sf[0][v];
      P[m * 32 + 16 + l16] = (__bf16)sf[1][v];
    }
    asm volatile("s_wait_dscnt 0" ::: "memory");   // wave-local LDS RAW
    __syncthreads();                               // Vt staged by all waves

    Frag16 pf;
    pf.q[0] = *reinterpret_cast<const U4*>(&P[l16 * 32 + half * 8]);
    pf.q[1] = *reinterpret_cast<const U4*>(&P[l16 * 32 + 16 + half * 8]);
#pragma unroll
    for (int ds = 0; ds < 8; ++ds) {
      Frag16 vf;
      int n = ds * 16 + l16;
      vf.q[0] = *reinterpret_cast<const U4*>(&Vt[n * 40 + half * 8]);
      vf.q[1] = *reinterpret_cast<const U4*>(&Vt[n * 40 + 16 + half * 8]);
      o[ds] = __builtin_amdgcn_wmma_f32_16x16x32_bf16(
          false, pf.v, false, vf.v, (short)0, o[ds], false, false);
    }
  }

  // normalize and store to [b][s][h*DH]
#pragma unroll
  for (int v = 0; v < 8; ++v) {
    float l = lacc[v];
#pragma unroll
    for (int off = 1; off < 16; off <<= 1) l += __shfl_xor(l, off, 32);
    float inv = 1.0f / l;
    int srow = qw + v + half * 8;
    __bf16* op = O + ((size_t)(b * SEQ + srow) * EMB) + h * DH;
#pragma unroll
    for (int ds = 0; ds < 8; ++ds)
      op[ds * 16 + l16] = (__bf16)(o[ds][v] * inv);
  }
}

// ---------------------------------------------------------------------------
extern "C" void kernel_launch(void* const* d_in, const int* in_sizes, int n_in,
                              void* d_out, int out_size, void* d_ws, size_t ws_size,
                              hipStream_t stream) {
  const float* x  = (const float*)d_in[0];
  const float* Wq = (const float*)d_in[1];
  const float* Wk = (const float*)d_in[2];
  const float* Wv = (const float*)d_in[3];
  const float* Wo = (const float*)d_in[4];
  const float* bo = (const float*)d_in[5];
  float* out = (float*)d_out;

  size_t off = 0;
  auto arena = [&](size_t bytes) -> void* {
    void* p = (char*)d_ws + off;
    off += (bytes + 255) & ~(size_t)255;
    return p;
  };
  __bf16* xb  = (__bf16*)arena((size_t)BB * SEQ * EMB * 2);
  __bf16* wqb = (__bf16*)arena((size_t)EMB * EMB * 2);
  __bf16* wkb = (__bf16*)arena((size_t)EMB * KV * 2);
  __bf16* wvb = (__bf16*)arena((size_t)EMB * KV * 2);
  __bf16* wob = (__bf16*)arena((size_t)EMB * EMB * 2);
  float*  Kf  = (float*)arena((size_t)BB * SEQ * KV * 4);
  float*  Vf  = (float*)arena((size_t)BB * SEQ * KV * 4);
  __bf16* Qhb = (__bf16*)arena((size_t)BB * SEQ * EMB * 2);
  __bf16* Kgb = (__bf16*)arena((size_t)BB * SEQ * KV * 2);
  __bf16* Vgb = (__bf16*)arena((size_t)BB * SEQ * KV * 2);
  __bf16* AOb = (__bf16*)arena((size_t)BB * SEQ * EMB * 2);
  float*  Qf  = out;   // d_out doubles as fp32 Q scratch; final GEMM overwrites

  const int Mr = BB * SEQ;                 // 4096
  dim3 blk128(128);

  cvt_bf16_kernel<<<2048, 256, 0, stream>>>(x,  xb,  (long)BB * SEQ * EMB);
  cvt_bf16_kernel<<<2048, 256, 0, stream>>>(Wq, wqb, (long)EMB * EMB);
  cvt_bf16_kernel<<<2048, 256, 0, stream>>>(Wk, wkb, (long)EMB * KV);
  cvt_bf16_kernel<<<2048, 256, 0, stream>>>(Wv, wvb, (long)EMB * KV);
  cvt_bf16_kernel<<<2048, 256, 0, stream>>>(Wo, wob, (long)EMB * EMB);

  gemm_bf16_wmma<<<dim3(EMB / 64, Mr / 64), blk128, 0, stream>>>(
      xb, wqb, nullptr, Qf, Mr, EMB, EMB);
  gemm_bf16_wmma<<<dim3(KV / 64, Mr / 64), blk128, 0, stream>>>(
      xb, wkb, nullptr, Kf, Mr, KV, EMB);
  gemm_bf16_wmma<<<dim3(KV / 64, Mr / 64), blk128, 0, stream>>>(
      xb, wvb, nullptr, Vf, Mr, KV, EMB);

  rope_pack_kernel<<<2048, 256, 0, stream>>>(Qf, Qhb, NH);
  rope_pack_kernel<<<2048, 256, 0, stream>>>(Kf, Kgb, NG);
  pack_v_kernel<<<2048, 256, 0, stream>>>(Vf, Vgb);

  attn_fwd_wmma<<<dim3(SEQ / 64, NH, BB), blk128, 0, stream>>>(Qhb, Kgb, Vgb, AOb);

  gemm_bf16_wmma<<<dim3(EMB / 64, Mr / 64), blk128, 0, stream>>>(
      AOb, wob, bo, out, Mr, EMB, EMB);
}
